// RDGCN_layer_68753836474514
// MI455X (gfx1250) — matrix operand order repaired and checked by reading, not compiled, not run on hardware
//
#include <hip/hip_runtime.h>
#include <hip/hip_bf16.h>

// Problem constants (from reference)
#define NN 50000
#define RR 500
#define DD 300
#define HH 600
#define EE 500000
#define ALPHA_C 0.1f
#define BETA_C 0.3f

typedef __attribute__((ext_vector_type(16))) __bf16 bf16x16;
typedef __attribute__((ext_vector_type(8)))  __bf16 bf16x8;
typedef __attribute__((ext_vector_type(4)))  __bf16 bf16x4;
typedef __attribute__((ext_vector_type(8)))  float  floatx8;

__device__ __forceinline__ float leaky(float x) { return x > 0.f ? x : 0.01f * x; }

// ---------------------------------------------------------------------------
// f32 -> packed bf16 conversion
// ---------------------------------------------------------------------------
__global__ __launch_bounds__(256)
void cvt_bf16_kernel(const float* __restrict__ in, __bf16* __restrict__ out, size_t n)
{
  size_t i = (size_t)blockIdx.x * blockDim.x + threadIdx.x;
  if (i < n) out[i] = (__bf16)in[i];
}

// ---------------------------------------------------------------------------
// LDS-staged WMMA bf16 GEMM.
//   C[m,n] = (sum_k Ae(m,k)*Be(k,n)) / (rowdiv[m]+1e-9)? + bias[n]?
//   Ae(m,k) = TRANSA ? A[k*lda+m] : A[m*lda+k]
//   Be(k,n) = TRANSB ? B[n*ldb+k] : B[k*ldb+n]
// 128 threads = 4 waves; workgroup computes a 32x32 C tile (one 16x16 per wave).
// LDS: As[m][k], BsT[n][k], pitch 80 bf16 (160 B: 16B-aligned rows, conflict-free
// transpose stores). Fragment reads are 16B ds_load_b128.
// Staging: full-tile fast path uses bf16x4 (global_load_b64) vector loads;
// edge tiles use branchless clamped loads (no exec-mask churn).
// Requires lda % 4 == 0 and ldb % 4 == 0 (holds: 500/300/600).
// ---------------------------------------------------------------------------
template<int TRANSA, int TRANSB>
__global__ __launch_bounds__(128)
void wmma_gemm_bf16(const __bf16* __restrict__ A, const __bf16* __restrict__ B,
                    float* __restrict__ C, int M, int Nc, int K,
                    int lda, int ldb, int ldc,
                    const float* __restrict__ rowdiv, const float* __restrict__ bias)
{
  constexpr int TM = 32, TN = 32, TK = 64;
  constexpr int PITCH = TK + 16;            // 80 elements = 160 bytes
  __shared__ __bf16 As[TM][PITCH];          // As[m][k]
  __shared__ __bf16 BsT[TN][PITCH];         // BsT[n][k]

  const int tid   = threadIdx.x;
  const int wave  = tid >> 5;
  const int lane  = tid & 31;
  const int l16   = lane & 15;
  const int half  = lane >> 4;
  const int wm    = (wave >> 1) * 16;       // wave's 16x16 sub-tile inside 32x32
  const int wn    = (wave & 1) * 16;
  const int tileM = blockIdx.x * TM;
  const int tileN = blockIdx.y * TN;

  const bool edgeM = (tileM + TM > M);
  const bool edgeN = (tileN + TN > Nc);

  floatx8 acc = {0.f,0.f,0.f,0.f,0.f,0.f,0.f,0.f};

  for (int k0 = 0; k0 < K; k0 += TK) {
    const bool fullK = (k0 + TK <= K);
    // ================= A staging =================
    if (TRANSA) {
      if (fullK && !edgeM) {
        // Ae(m,k)=A[k*lda+m]; vector across m, transpose-store into As[m][k]
        #pragma unroll
        for (int j = 0; j < 4; ++j) {
          int vid = tid + 128 * j;
          int k  = vid >> 3;            // 0..63
          int ms = (vid & 7) * 4;       // 0,4,..,28
          bf16x4 v = *(const bf16x4*)&A[(size_t)(k0 + k) * lda + tileM + ms];
          As[ms + 0][k] = v[0];
          As[ms + 1][k] = v[1];
          As[ms + 2][k] = v[2];
          As[ms + 3][k] = v[3];
        }
      } else {
        #pragma unroll
        for (int i = tid; i < TM * TK; i += 128) {
          int m = i & (TM - 1), k = i >> 5;
          int gm = tileM + m, gk = k0 + k;
          bool ok = (gm < M) && (gk < K);
          size_t idx = ok ? ((size_t)gk * lda + gm) : 0;
          __bf16 v = A[idx];
          As[m][k] = ok ? v : (__bf16)0.f;
        }
      }
    } else {
      if (fullK && !edgeM) {
        // Ae(m,k)=A[m*lda+k]; vector across k, vector LDS store
        #pragma unroll
        for (int j = 0; j < 4; ++j) {
          int vid = tid + 128 * j;
          int m  = vid >> 4;            // 0..31
          int ks = (vid & 15) * 4;      // 0..60
          bf16x4 v = *(const bf16x4*)&A[(size_t)(tileM + m) * lda + k0 + ks];
          *(bf16x4*)&As[m][ks] = v;
        }
      } else {
        #pragma unroll
        for (int i = tid; i < TM * TK; i += 128) {
          int k = i & (TK - 1), m = i >> 6;
          int gm = tileM + m, gk = k0 + k;
          bool ok = (gm < M) && (gk < K);
          size_t idx = ok ? ((size_t)gm * lda + gk) : 0;
          __bf16 v = A[idx];
          As[m][k] = ok ? v : (__bf16)0.f;
        }
      }
    }
    // ================= B staging =================
    if (TRANSB) {
      if (fullK && !edgeN) {
        // Be(k,n)=B[n*ldb+k]; vector across k, vector LDS store
        #pragma unroll
        for (int j = 0; j < 4; ++j) {
          int vid = tid + 128 * j;
          int n  = vid >> 4;
          int ks = (vid & 15) * 4;
          bf16x4 v = *(const bf16x4*)&B[(size_t)(tileN + n) * ldb + k0 + ks];
          *(bf16x4*)&BsT[n][ks] = v;
        }
      } else {
        #pragma unroll
        for (int i = tid; i < TK * TN; i += 128) {
          int k = i & (TK - 1), n = i >> 6;
          int gn = tileN + n, gk = k0 + k;
          bool ok = (gn < Nc) && (gk < K);
          size_t idx = ok ? ((size_t)gn * ldb + gk) : 0;
          __bf16 v = B[idx];
          BsT[n][k] = ok ? v : (__bf16)0.f;
        }
      }
    } else {
      if (fullK && !edgeN) {
        // Be(k,n)=B[k*ldb+n]; vector across n, transpose-store into BsT[n][k]
        #pragma unroll
        for (int j = 0; j < 4; ++j) {
          int vid = tid + 128 * j;
          int k  = vid >> 3;
          int ns = (vid & 7) * 4;
          bf16x4 v = *(const bf16x4*)&B[(size_t)(k0 + k) * ldb + tileN + ns];
          BsT[ns + 0][k] = v[0];
          BsT[ns + 1][k] = v[1];
          BsT[ns + 2][k] = v[2];
          BsT[ns + 3][k] = v[3];
        }
      } else {
        #pragma unroll
        for (int i = tid; i < TK * TN; i += 128) {
          int n = i & (TN - 1), k = i >> 5;
          int gn = tileN + n, gk = k0 + k;
          bool ok = (gn < Nc) && (gk < K);
          size_t idx = ok ? ((size_t)gk * ldb + gn) : 0;
          __bf16 v = B[idx];
          BsT[n][k] = ok ? v : (__bf16)0.f;
        }
      }
    }
    // Prefetch next K slab (global_prefetch_b8)
    if (k0 + TK < K) {
      int gkn = k0 + TK + (TRANSA ? (tid >> 5) : (tid & (TK - 1)));
      if (gkn < K) {
        const __bf16* pa = TRANSA ? &A[(size_t)gkn * lda + tileM]
                                  : &A[(size_t)(tileM + (tid >> 6)) * lda + gkn];
        __builtin_prefetch(pa, 0, 1);
      }
    }
    __syncthreads();

    // ---- two 16x16x32 WMMAs; all fragment reads are 16B ds_load_b128 ----
    #pragma unroll
    for (int kk = 0; kk < TK; kk += 32) {
      const __bf16* arow = &As[wm + l16][kk];
      bf16x8 alo = *(const bf16x8*)(arow + half * 8);
      bf16x8 ahi = *(const bf16x8*)(arow + 16 + half * 8);
      const __bf16* brow = &BsT[wn + l16][kk];
      bf16x8 blo = *(const bf16x8*)(brow + half * 8);
      bf16x8 bhi = *(const bf16x8*)(brow + 16 + half * 8);
      bf16x16 a = __builtin_shufflevector(alo, ahi, 0,1,2,3,4,5,6,7,8,9,10,11,12,13,14,15);
      bf16x16 b = __builtin_shufflevector(blo, bhi, 0,1,2,3,4,5,6,7,8,9,10,11,12,13,14,15);
      acc = __builtin_amdgcn_wmma_f32_16x16x32_bf16(
          false, a, false, b, (short)0, acc, false, false);
    }
    __syncthreads();
  }

  // ---- epilogue: C/D layout VGPR v -> M = half*8 + v, N = l16 ----
  #pragma unroll
  for (int v = 0; v < 8; ++v) {
    int m = tileM + wm + half * 8 + v;
    int n = tileN + wn + l16;
    if (m < M && n < Nc) {
      float val = acc[v];
      if (rowdiv) val = val / (rowdiv[m] + 1e-9f);
      if (bias)   val += bias[n];
      C[(size_t)m * ldc + n] = val;
    }
  }
}

// ---------------------------------------------------------------------------
// colsum[c] += partial sums of Mat[:,c]
// ---------------------------------------------------------------------------
__global__ __launch_bounds__(256)
void colsum_kernel(const float* __restrict__ Mat, float* __restrict__ out,
                   int nrows, int ncols)
{
  int c = blockIdx.x * blockDim.x + threadIdx.x;
  if (c >= ncols) return;
  float s = 0.f;
  for (int r = blockIdx.y; r < nrows; r += gridDim.y)
    s += Mat[(size_t)r * ncols + c];
  atomicAdd(&out[c], s);
}

// ---------------------------------------------------------------------------
// f[r] = dot(Hm[r,:], a) + bscalar[0]
// ---------------------------------------------------------------------------
__global__ __launch_bounds__(256)
void rowdot_kernel(const float* __restrict__ Hm, const float* __restrict__ a,
                   const float* __restrict__ bscalar, float* __restrict__ f,
                   int Hc)
{
  __shared__ float red[256];
  int r = blockIdx.x;
  float s = 0.f;
  for (int j = threadIdx.x; j < Hc; j += blockDim.x)
    s += Hm[(size_t)r * Hc + j] * a[j];
  red[threadIdx.x] = s; __syncthreads();
  for (int off = 128; off > 0; off >>= 1) {
    if (threadIdx.x < off) red[threadIdx.x] += red[threadIdx.x + off];
    __syncthreads();
  }
  if (threadIdx.x == 0) f[r] = red[0] + (bscalar ? bscalar[0] : 0.f);
}

// ---------------------------------------------------------------------------
// Dense attention row
// ---------------------------------------------------------------------------
__global__ __launch_bounds__(256)
void att_row_kernel(const float* __restrict__ f1, const float* __restrict__ f2,
                    const float* __restrict__ adj, const float* __restrict__ val,
                    float* __restrict__ out, int Hc)
{
  __shared__ float coef[RR];
  __shared__ float red[256];
  int i = blockIdx.x;
  float fi = f1[i];
  float lmax = -1e30f;
  for (int j = threadIdx.x; j < RR; j += blockDim.x) {
    float z = leaky(fi + f2[j]);
    if (adj[(size_t)i * RR + j] <= 0.f) z += -1e9f;
    coef[j] = z;
    lmax = fmaxf(lmax, z);
  }
  red[threadIdx.x] = lmax; __syncthreads();
  for (int off = 128; off > 0; off >>= 1) {
    if (threadIdx.x < off) red[threadIdx.x] = fmaxf(red[threadIdx.x], red[threadIdx.x + off]);
    __syncthreads();
  }
  lmax = red[0]; __syncthreads();

  float lsum = 0.f;
  for (int j = threadIdx.x; j < RR; j += blockDim.x) {
    float e = __expf(coef[j] - lmax);
    coef[j] = e;
    lsum += e;
  }
  red[threadIdx.x] = lsum; __syncthreads();
  for (int off = 128; off > 0; off >>= 1) {
    if (threadIdx.x < off) red[threadIdx.x] += red[threadIdx.x + off];
    __syncthreads();
  }
  float inv = 1.f / red[0]; __syncthreads();

  for (int h = threadIdx.x; h < Hc; h += blockDim.x) {
    float s = 0.f;
    for (int j = 0; j < RR; ++j) s += coef[j] * val[(size_t)j * Hc + h];
    float o = s * inv;
    out[(size_t)i * Hc + h] = o > 0.f ? o : 0.f;
  }
}

// ---------------------------------------------------------------------------
// Global-softmax over edges helpers
// ---------------------------------------------------------------------------
__global__ __launch_bounds__(512)
void tmax_kernel(const float* __restrict__ t, float* __restrict__ gmax)
{
  __shared__ float red[512];
  float m = -1e30f;
  for (int r = threadIdx.x; r < RR; r += blockDim.x) m = fmaxf(m, leaky(t[r]));
  red[threadIdx.x] = m; __syncthreads();
  for (int off = 256; off > 0; off >>= 1) {
    if (threadIdx.x < off) red[threadIdx.x] = fmaxf(red[threadIdx.x], red[threadIdx.x + off]);
    __syncthreads();
  }
  if (threadIdx.x == 0) gmax[0] = red[0];
}

__global__ __launch_bounds__(256)
void edge_denom_kernel(const float* __restrict__ t, const int* __restrict__ rvals,
                       const float* __restrict__ gmax, float* __restrict__ denom)
{
  __shared__ float red[256];
  float gm = gmax[0];
  float s = 0.f;
  int stride = gridDim.x * blockDim.x;
  for (int e = blockIdx.x * blockDim.x + threadIdx.x; e < EE; e += stride)
    s += __expf(leaky(t[rvals[e]]) - gm);
  red[threadIdx.x] = s; __syncthreads();
  for (int off = 128; off > 0; off >>= 1) {
    if (threadIdx.x < off) red[threadIdx.x] += red[threadIdx.x + off];
    __syncthreads();
  }
  if (threadIdx.x == 0) atomicAdd(denom, red[0]);
}

__global__ __launch_bounds__(128)
void edge_scatter_kernel(const float* __restrict__ X, const int* __restrict__ ridx,
                         const int* __restrict__ rvals, const float* __restrict__ t,
                         const float* __restrict__ gmax, const float* __restrict__ denom,
                         float* __restrict__ out)
{
  int e = blockIdx.x;
  int dst = ridx[e];
  int src = ridx[EE + e];
  float c = __expf(leaky(t[rvals[e]]) - gmax[0]) / denom[0];
  const float* xs = X + (size_t)src * DD;
  float* od = out + (size_t)dst * DD;
  for (int d = threadIdx.x; d < DD; d += blockDim.x)
    atomicAdd(&od[d], c * xs[d]);
}

// ---------------------------------------------------------------------------
// Diag GCN scatter: out[dst,:] += mval * X[src,:] * w0[:]
// ---------------------------------------------------------------------------
__global__ __launch_bounds__(128)
void gcn_scatter_kernel(const float* __restrict__ X, const float* __restrict__ w0,
                        const int* __restrict__ midx, const float* __restrict__ mvals,
                        float* __restrict__ out)
{
  int e = blockIdx.x;
  int dst = midx[e];
  int src = midx[EE + e];
  float c = mvals[e];
  const float* xs = X + (size_t)src * DD;
  float* od = out + (size_t)dst * DD;
  for (int d = threadIdx.x; d < DD; d += blockDim.x)
    atomicAdd(&od[d], c * xs[d] * w0[d]);
}

// ---------------------------------------------------------------------------
// out = X0 + alpha * relu(Hs)
// ---------------------------------------------------------------------------
__global__ __launch_bounds__(256)
void axpy_relu_kernel(const float* __restrict__ X0, const float* __restrict__ Hs,
                      float* __restrict__ out, float alpha, size_t n)
{
  size_t i = (size_t)blockIdx.x * blockDim.x + threadIdx.x;
  if (i < n) {
    float h = Hs[i];
    out[i] = X0[i] + alpha * (h > 0.f ? h : 0.f);
  }
}

// ---------------------------------------------------------------------------
// Highway: t=sigmoid(g); l2=relu(agg); out = t*l2 + (1-t)*l1
// ---------------------------------------------------------------------------
__global__ __launch_bounds__(256)
void highway_kernel(const float* __restrict__ l1, const float* __restrict__ agg,
                    const float* __restrict__ g, float* __restrict__ out, size_t n)
{
  size_t i = (size_t)blockIdx.x * blockDim.x + threadIdx.x;
  if (i < n) {
    float t  = 1.f / (1.f + __expf(-g[i]));
    float l2 = fmaxf(agg[i], 0.f);
    out[i] = t * l2 + (1.f - t) * l1[i];
  }
}

// ===========================================================================
extern "C" void kernel_launch(void* const* d_in, const int* in_sizes, int n_in,
                              void* d_out, int out_size, void* d_ws, size_t ws_size,
                              hipStream_t stream)
{
  const float* X0     = (const float*)d_in[0];
  const float* head   = (const float*)d_in[1];
  const float* tail   = (const float*)d_in[2];
  const float* adj    = (const float*)d_in[3];
  const int*   r_idx  = (const int*)d_in[4];
  const int*   r_vals = (const int*)d_in[5];
  const int*   m_idx  = (const int*)d_in[6];
  const float* m_vals = (const float*)d_in[7];
  const float* W_s    = (const float*)d_in[8];
  const float* a1_s   = (const float*)d_in[9];
  const float* b1_s   = (const float*)d_in[10];
  const float* a2_s   = (const float*)d_in[11];
  const float* b2_s   = (const float*)d_in[12];
  const float* w_sp1  = (const float*)d_in[13];
  const float* b_sp1  = (const float*)d_in[14];
  const float* w_sp2  = (const float*)d_in[15];
  const float* b_sp2  = (const float*)d_in[16];
  const float* W_d    = (const float*)d_in[17];
  const float* b_d    = (const float*)d_in[18];
  const float* a1_d   = (const float*)d_in[19];
  const float* b1_d   = (const float*)d_in[20];
  const float* a2_d   = (const float*)d_in[21];
  const float* b2_d   = (const float*)d_in[22];
  const float* w0_1   = (const float*)d_in[23];
  const float* w0_2   = (const float*)d_in[24];
  const float* Wg1    = (const float*)d_in[25];
  const float* bg1    = (const float*)d_in[26];
  const float* Wg2    = (const float*)d_in[27];
  const float* bg2    = (const float*)d_in[28];

  const size_t ND = (size_t)NN * DD;
  const size_t RH = (size_t)RR * HH;
  const size_t NR = (size_t)NN * RR;

  float* out_X  = (float*)d_out;       // [N, D]
  float* out_H3 = out_X + ND;          // [R, H]

  // ---- workspace carve-out (byte-based, 16B aligned) ----
  char* wsb_base = (char*)d_ws;
  size_t off = 0;
  auto alloc = [&](size_t bytes) {
    char* p = wsb_base + off;
    off += (bytes + 15) & ~(size_t)15;
    return (void*)p;
  };
  float*  cs_h   = (float*)alloc(RR * sizeof(float));
  float*  cs_t   = (float*)alloc(RR * sizeof(float));
  float*  dualX  = (float*)alloc(RH * sizeof(float));
  float*  htmp   = (float*)alloc(RH * sizeof(float));
  float*  dualH1 = (float*)alloc(RH * sizeof(float));
  float*  dualH2 = (float*)alloc(RH * sizeof(float));
  float*  f1     = (float*)alloc(RR * sizeof(float));
  float*  f2     = (float*)alloc(RR * sizeof(float));
  float*  tr     = (float*)alloc(RR * sizeof(float));
  float*  gmax   = (float*)alloc(sizeof(float));
  float*  gden   = (float*)alloc(sizeof(float));
  float*  scat   = (float*)alloc(ND * sizeof(float));
  float*  pX1    = (float*)alloc(ND * sizeof(float));
  float*  pX2    = (float*)alloc(ND * sizeof(float));
  float*  gbuf   = (float*)alloc(ND * sizeof(float));
  float*  gcn1   = (float*)alloc(ND * sizeof(float));
  __bf16* headb  = (__bf16*)alloc(NR * sizeof(__bf16));
  __bf16* tailb  = (__bf16*)alloc(NR * sizeof(__bf16));
  __bf16* xb     = (__bf16*)alloc(ND * sizeof(__bf16));   // reused for every N*D operand
  __bf16* featb  = (__bf16*)alloc(RH * sizeof(__bf16));   // reused per dense_att
  __bf16* wsb    = (__bf16*)alloc((size_t)HH * HH * sizeof(__bf16));
  __bf16* wdb    = (__bf16*)alloc((size_t)HH * HH * sizeof(__bf16));
  __bf16* wg1b   = (__bf16*)alloc((size_t)DD * DD * sizeof(__bf16));
  __bf16* wg2b   = (__bf16*)alloc((size_t)DD * DD * sizeof(__bf16));

  const int nd_blocks = (int)((ND + 255) / 256);
  auto cvt = [&](const float* src, __bf16* dst, size_t n) {
    cvt_bf16_kernel<<<(int)((n + 255) / 256), 256, 0, stream>>>(src, dst, n);
  };

  dim3 grid_rd((RR + 31) / 32, (DD + 31) / 32);   // compute_r tiles  (16, 10)
  dim3 grid_rh((RR + 31) / 32, (HH + 31) / 32);   // feat @ W.T tiles (16, 19)
  dim3 grid_ndd((NN + 31) / 32, (DD + 31) / 32);  // highway gate tiles

  // ---- one-time conversions (weights + masks are layer constants) ----
  cvt(head, headb, NR);
  cvt(tail, tailb, NR);
  cvt(W_s, wsb, (size_t)HH * HH);
  cvt(W_d, wdb, (size_t)HH * HH);
  cvt(Wg1, wg1b, (size_t)DD * DD);
  cvt(Wg2, wg2b, (size_t)DD * DD);

  // ---- column sums of head_r / tail_r ----
  hipMemsetAsync(cs_h, 0, RR * sizeof(float), stream);
  hipMemsetAsync(cs_t, 0, RR * sizeof(float), stream);
  colsum_kernel<<<dim3((RR + 255) / 256, 64), 256, 0, stream>>>(head, cs_h, NN, RR);
  colsum_kernel<<<dim3((RR + 255) / 256, 64), 256, 0, stream>>>(tail, cs_t, NN, RR);

  auto compute_r = [&](const float* X, float* dX) {
    cvt(X, xb, ND);
    // L  = head.T @ X / cs_h  -> dX[:, 0:D]     (A trans, lda=R; B normal, ldb=D)
    wmma_gemm_bf16<1, 0><<<grid_rd, 128, 0, stream>>>(headb, xb, dX, RR, DD, NN,
                                                      RR, DD, HH, cs_h, nullptr);
    // Rm = tail.T @ X / cs_t  -> dX[:, D:2D]
    wmma_gemm_bf16<1, 0><<<grid_rd, 128, 0, stream>>>(tailb, xb, dX + DD, RR, DD, NN,
                                                      RR, DD, HH, cs_t, nullptr);
  };

  auto dense_att = [&](const float* feat, const float* val, const __bf16* Wb,
                       const float* bW, const float* a1, const float* b1,
                       const float* a2, const float* b2, float* outH) {
    cvt(feat, featb, RH);
    // h = feat @ W.T (+ bW):  M=R, N=H, K=H ; B(k,n)=W[n*H+k] -> TRANSB=1
    wmma_gemm_bf16<0, 1><<<grid_rh, 128, 0, stream>>>(featb, Wb, htmp, RR, HH, HH,
                                                      HH, HH, HH, nullptr, bW);
    rowdot_kernel<<<RR, 256, 0, stream>>>(htmp, a1, b1, f1, HH);
    rowdot_kernel<<<RR, 256, 0, stream>>>(htmp, a2, b2, f2, HH);
    att_row_kernel<<<RR, 256, 0, stream>>>(f1, f2, adj, val, outH, HH);
  };

  auto sparse_att = [&](const float* X, const float* dualH, const float* w,
                        const float* bs, float* scatbuf) {
    rowdot_kernel<<<RR, 256, 0, stream>>>(dualH, w, bs, tr, HH);
    tmax_kernel<<<1, 512, 0, stream>>>(tr, gmax);
    hipMemsetAsync(gden, 0, sizeof(float), stream);
    edge_denom_kernel<<<256, 256, 0, stream>>>(tr, r_vals, gmax, gden);
    hipMemsetAsync(scatbuf, 0, ND * sizeof(float), stream);
    edge_scatter_kernel<<<EE, 128, 0, stream>>>(X, r_idx, r_vals, tr, gmax, gden, scatbuf);
  };

  // ---- layer pipeline ----
  compute_r(X0, dualX);                                                   // dual_X_1
  dense_att(dualX, dualX, wsb, nullptr, a1_s, b1_s, a2_s, b2_s, dualH1);  // dual_H_1
  sparse_att(X0, dualH1, w_sp1, b_sp1, scat);                             // primal_H_1
  axpy_relu_kernel<<<nd_blocks, 256, 0, stream>>>(X0, scat, pX1, ALPHA_C, ND);

  compute_r(pX1, dualX);                                                  // dual_X_2
  dense_att(dualX, dualH1, wdb, b_d, a1_d, b1_d, a2_d, b2_d, dualH2);     // dual_H_2
  sparse_att(pX1, dualH2, w_sp2, b_sp2, scat);                            // primal_H_2
  axpy_relu_kernel<<<nd_blocks, 256, 0, stream>>>(X0, scat, pX2, BETA_C, ND);

  // diag GCN 1 + highway 1
  hipMemsetAsync(scat, 0, ND * sizeof(float), stream);
  gcn_scatter_kernel<<<EE, 128, 0, stream>>>(pX2, w0_1, m_idx, m_vals, scat);
  cvt(pX2, xb, ND);
  wmma_gemm_bf16<0, 0><<<grid_ndd, 128, 0, stream>>>(xb, wg1b, gbuf, NN, DD, DD,
                                                     DD, DD, DD, nullptr, bg1);
  highway_kernel<<<nd_blocks, 256, 0, stream>>>(pX2, scat, gbuf, gcn1, ND);

  // diag GCN 2 + highway 2 -> output
  hipMemsetAsync(scat, 0, ND * sizeof(float), stream);
  gcn_scatter_kernel<<<EE, 128, 0, stream>>>(gcn1, w0_2, m_idx, m_vals, scat);
  cvt(gcn1, xb, ND);
  wmma_gemm_bf16<0, 0><<<grid_ndd, 128, 0, stream>>>(xb, wg2b, gbuf, NN, DD, DD,
                                                     DD, DD, DD, nullptr, bg2);
  highway_kernel<<<nd_blocks, 256, 0, stream>>>(gcn1, scat, gbuf, out_X, ND);

  // dual_H_3
  compute_r(out_X, dualX);                                                // dual_X_3
  dense_att(dualX, dualH2, wdb, b_d, a1_d, b1_d, a2_d, b2_d, out_H3);     // dual_H_3
}
